// Network_4964982194321
// MI455X (gfx1250) — matrix operand order repaired and checked by reading, not compiled, not run on hardware
//
#include <hip/hip_runtime.h>
#include <hip/hip_bf16.h>
#include <math.h>

// ---------------------------------------------------------------------------
// BrainGNN-style forward pass for MI455X (gfx1250).
// All GEMMs run through one wave32-per-tile WMMA kernel whose inner loop is
// 4x global_load_b128 + v_wmma_f32_16x16x32_bf16 (operands pre-packed as bf16
// in fragment-friendly zero-padded layouts, so no guards / converts inline).
// ---------------------------------------------------------------------------

typedef __attribute__((ext_vector_type(16))) __bf16 v16bf;
typedef __attribute__((ext_vector_type(8)))  float  v8f;
typedef __attribute__((ext_vector_type(4)))  unsigned int u32x4;

namespace {
constexpr int BB    = 64;    // graphs
constexpr int RR    = 200;   // nodes / graph
constexpr int INDIM = 200;
constexpr int DEG   = 32;
constexpr int KCOMM = 8;
constexpr int D1 = 32, D2 = 32, D3 = 512;
constexpr int KP1 = 100, KP2 = 50;       // kept after pool1 / pool2
constexpr int NNODES = BB * RR;          // 12800
constexpr int NCOLS  = (KCOMM + 1) * 32; // 288
// padded GEMM dims
constexpr int KPAD200 = 224;  // K=200 -> mult of 32
constexpr int MPAD200 = 208;  // M=200 -> mult of 16
constexpr int KPAD100 = 128;  // K=100 -> mult of 32
constexpr int NPAD100 = 112;  // M/N=100 -> mult of 16
}

__device__ __forceinline__ unsigned short f2bf(float f) {
  union { float f; unsigned u; } v; v.f = f;
  unsigned u = v.u;
  unsigned r = u + 0x7FFFu + ((u >> 16) & 1u);   // round-to-nearest-even
  return (unsigned short)(r >> 16);
}

// ---------------------------------------------------------------------------
// C[b] = A[b] * B[b] (+ bias).  Apk: bf16 [batch, Mp, Kp] row-major (padded,
// zero-filled).  Bt: bf16 [batch, Np, Kp] = B transposed (padded).  One wave
// per 16x16 C tile; fragments per CDNA5 ISA 7.12.2:
//   A lane(l) half=l>>4: VGPR0-3 = k half*8+0..7, VGPR4-7 = k 16+half*8+0..7
//   B lane(l):           VGPR0-3 = k half*8+0..7, VGPR4-7 = k 16+half*8+0..7 of column n0+(l&15)
// Both are contiguous bf16 runs in the packed layouts -> b128 loads.
// ---------------------------------------------------------------------------
__global__ __launch_bounds__(32)
void wmma_gemm_pk(const unsigned short* __restrict__ Apk,
                  const unsigned short* __restrict__ Bt,
                  float* __restrict__ C, const float* __restrict__ bias,
                  int M, int N, int Kp, int ldc,
                  long long sA, long long sB, long long sC)
{
  const int lane = threadIdx.x & 31;
  const int half = lane >> 4;
  const int l15  = lane & 15;
  const long long bz = blockIdx.z;
  const int m0 = blockIdx.y * 16;
  const int n0 = blockIdx.x * 16;
  const int mrow = m0 + l15;
  const int ncol = n0 + l15;

  const unsigned short* ap = Apk + bz * sA + (long long)mrow * Kp + (half << 3);
  const unsigned short* bp = Bt  + bz * sB + (long long)ncol * Kp + (half << 3);

  v8f acc = {};
  for (int k0 = 0; k0 < Kp; k0 += 32) {
    union { v16bf v; u32x4 u[2]; } av, bv;
    av.u[0] = *(const u32x4*)(ap);        // k = half*8 + 0..7
    av.u[1] = *(const u32x4*)(ap + 16);   // k = 16 + half*8 + 0..7
    bv.u[0] = *(const u32x4*)(bp);        // k = half*8 + 0..7
    bv.u[1] = *(const u32x4*)(bp + 16);   // k = 16 + half*8 + 0..7
    acc = __builtin_amdgcn_wmma_f32_16x16x32_bf16(false, av.v, false, bv.v,
                                                  (short)0, acc, false, false);
    ap += 32; bp += 32;
  }

  float badd = 0.0f;
  if (bias != nullptr && ncol < N) badd = bias[ncol];
  C += bz * sC;
#pragma unroll
  for (int r = 0; r < 8; ++r) {           // C/D: M = half*8 + r, N = l15
    const int m = m0 + (half << 3) + r;
    if (m < M && ncol < N) C[(long long)m * ldc + ncol] = acc[r] + badd;
  }
}

// --------------------------- pack kernels ----------------------------------

// dst[b, m, k] = bf16(src[b, m*lda + k]), zero outside [M,K)
__global__ void k_packA(const float* __restrict__ src, unsigned short* __restrict__ dst,
                        int M, int K, int lda, int Mp, int Kp,
                        long long sS, long long sD) {
  int t = blockIdx.x * blockDim.x + threadIdx.x;
  if (t >= Mp * Kp) return;
  const int m = t / Kp, k = t % Kp;
  const float* S = src + (long long)blockIdx.z * sS;
  unsigned short* D = dst + (long long)blockIdx.z * sD;
  const float v = (m < M && k < K) ? S[(long long)m * lda + k] : 0.0f;
  D[t] = f2bf(v);
}

// dst[b, n, k] = bf16(src[b, k*ldb + n]), zero outside [N,K)  (B transposed)
__global__ void k_packBT(const float* __restrict__ src, unsigned short* __restrict__ dst,
                         int K, int N, int ldb, int Np, int Kp,
                         long long sS, long long sD) {
  int t = blockIdx.x * blockDim.x + threadIdx.x;
  if (t >= Np * Kp) return;
  const int n = t / Kp, k = t % Kp;
  const float* S = src + (long long)blockIdx.z * sS;
  unsigned short* D = dst + (long long)blockIdx.z * sD;
  const float v = (n < N && k < K) ? S[(long long)k * ldb + n] : 0.0f;
  D[t] = f2bf(v);
}

// --------------------------- support kernels -------------------------------

__global__ void k_zero(float* p, long long n) {
  long long t = (long long)blockIdx.x * blockDim.x + threadIdx.x;
  if (t < n) p[t] = 0.0f;
}

// Deterministic dense adjacency: edges are grouped by dst (DEG each), one
// thread owns one dst row -> no atomics, fixed accumulation order.
__global__ void k_build_adj(const int* __restrict__ ei, const float* __restrict__ ea,
                            float* __restrict__ Aa, float* __restrict__ Mm) {
  int n = blockIdx.x * blockDim.x + threadIdx.x;   // dst node id
  if (n >= NNODES) return;
  const int g = n / RR, i = n % RR;
  float* Ar = Aa + ((long long)g * RR + i) * RR;
  float* Mr = Mm + ((long long)g * RR + i) * RR;
  for (int e = n * DEG; e < n * DEG + DEG; ++e) {
    const int j = ei[e] % RR;        // row 0 of edge_index = src
    Ar[j] += ea[e];
    Mr[j] = 1.0f;
  }
}

// W2r[i, c*32+o] = nn_w2[c, i*32+o] for c<8 ; W2r[i, 256+o] = nn_b2[i*32+o]
__global__ void k_w2r(const float* __restrict__ w2, const float* __restrict__ b2,
                      float* __restrict__ W2r, int Kin) {
  int t = blockIdx.x * blockDim.x + threadIdx.x;
  if (t >= Kin * NCOLS) return;
  const int i = t / NCOLS, col = t % NCOLS;
  const int c = col / 32, o = col % 32;
  W2r[t] = (c < KCOMM) ? w2[(long long)c * Kin * 32 + i * 32 + o] : b2[i * 32 + o];
}

// wpos[n,c] = relu(sum_j pos[n,j] * W1[j,c])
__global__ void k_posw(const float* __restrict__ pos, const float* __restrict__ W1,
                       float* __restrict__ out, int nodes, int pdim) {
  int n = blockIdx.x * blockDim.x + threadIdx.x;
  if (n >= nodes) return;
  float acc[KCOMM];
#pragma unroll
  for (int c = 0; c < KCOMM; ++c) acc[c] = 0.0f;
  const float* pr = pos + (long long)n * pdim;
  for (int j = 0; j < pdim; ++j) {
    const float p = pr[j];
#pragma unroll
    for (int c = 0; c < KCOMM; ++c) acc[c] += p * W1[j * KCOMM + c];
  }
#pragma unroll
  for (int c = 0; c < KCOMM; ++c) out[n * KCOMM + c] = fmaxf(acc[c], 0.0f);
}

// h[n,o] = sum_c wpos[n,c] * T[n, c*32+o] + T[n, 256+o]
__global__ void k_combine(const float* __restrict__ T, const float* __restrict__ wpos,
                          float* __restrict__ h, int nodes) {
  int t = blockIdx.x * blockDim.x + threadIdx.x;
  if (t >= nodes * 32) return;
  const int n = t / 32, o = t % 32;
  const float* Tn = T + (long long)n * NCOLS;
  float acc = Tn[KCOMM * 32 + o];
#pragma unroll
  for (int c = 0; c < KCOMM; ++c) acc += wpos[n * KCOMM + c] * Tn[c * 32 + o];
  h[t] = acc;
}

// Row softmax of Aw with self-loops, emitted directly as packed bf16 A-layout
// [BB, Mp, Kp] (rows >= Kdim and cols >= Kdim zero-filled).
__global__ void k_softmax_pk(const float* __restrict__ A, const float* __restrict__ Mk,
                             unsigned short* __restrict__ opk, int Kdim, int Mp, int Kp) {
  long long row = (long long)blockIdx.x * blockDim.x + threadIdx.x;
  if (row >= (long long)BB * Mp) return;
  const int b = (int)(row / Mp), i = (int)(row % Mp);
  unsigned short* Or = opk + ((long long)b * Mp + i) * Kp;
  if (i >= Kdim) {
    for (int j = 0; j < Kp; ++j) Or[j] = 0;
    return;
  }
  const float* Ar = A  + ((long long)b * Kdim + i) * Kdim;
  const float* Mr = Mk + ((long long)b * Kdim + i) * Kdim;
  float mx = -1e30f;
  for (int j = 0; j < Kdim; ++j) {
    const bool valid = (Mr[j] != 0.0f) || (j == i);
    const float v = (j == i) ? 1.0f : Ar[j];
    if (valid && v > mx) mx = v;
  }
  float s = 0.0f;
  for (int j = 0; j < Kdim; ++j) {
    const bool valid = (Mr[j] != 0.0f) || (j == i);
    const float v = (j == i) ? 1.0f : Ar[j];
    s += valid ? __expf(v - mx) : 0.0f;
  }
  const float inv = 1.0f / s;
  for (int j = 0; j < Kdim; ++j) {
    const bool valid = (Mr[j] != 0.0f) || (j == i);
    const float v = (j == i) ? 1.0f : Ar[j];
    const float e = valid ? __expf(v - mx) * inv : 0.0f;
    Or[j] = f2bf(e);
  }
  for (int j = Kdim; j < Kp; ++j) Or[j] = 0;
}

// score[n] = sigmoid(dot(x[n,:32], w) / ||w||)
__global__ void k_score(const float* __restrict__ x, const float* __restrict__ w,
                        float* __restrict__ score, int nodes) {
  int n = blockIdx.x * blockDim.x + threadIdx.x;
  if (n >= nodes) return;
  float nrm = 0.0f, d = 0.0f;
#pragma unroll
  for (int o = 0; o < 32; ++o) nrm += w[o] * w[o];
  nrm = sqrtf(nrm);
#pragma unroll
  for (int o = 0; o < 32; ++o) d += x[(long long)n * 32 + o] * w[o];
  score[n] = 1.0f / (1.0f + __expf(-d / nrm));
}

// Per-graph descending top-K with first-index tie-break (torch.topk order).
__global__ __launch_bounds__(256)
void k_topk(const float* __restrict__ score, int* __restrict__ idx,
            float* __restrict__ sc, int Kin, int Kkeep) {
  __shared__ float svals[256];
  __shared__ int   sidx[256];
  __shared__ unsigned char used[256];
  const int b = blockIdx.x, t = threadIdx.x;
  const float* s = score + (long long)b * Kin;
  used[t] = 0;
  __syncthreads();
  for (int kk = 0; kk < Kkeep; ++kk) {
    float best = -1e30f; int bi = -1;
    for (int j = t; j < Kin; j += 256) {
      if (!used[j] && (s[j] > best || (s[j] == best && (bi < 0 || j < bi)))) {
        best = s[j]; bi = j;
      }
    }
    svals[t] = best; sidx[t] = bi;
    __syncthreads();
    for (int off = 128; off > 0; off >>= 1) {
      if (t < off) {
        const bool take = (svals[t + off] > svals[t]) ||
                          (svals[t + off] == svals[t] && sidx[t + off] >= 0 &&
                           (sidx[t] < 0 || sidx[t + off] < sidx[t]));
        if (take) { svals[t] = svals[t + off]; sidx[t] = sidx[t + off]; }
      }
      __syncthreads();
    }
    if (t == 0) {
      idx[b * Kkeep + kk] = sidx[0];
      sc [b * Kkeep + kk] = svals[0];
      used[sidx[0]] = 1;
    }
    __syncthreads();
  }
}

// Gather pooled nodes: xk = x[idx]*sc ; sub-select A,M ; gather pos ; sigmoid(sc).
__global__ void k_gather(const float* __restrict__ x, const float* __restrict__ A,
                         const float* __restrict__ Mk, const float* __restrict__ pos,
                         const int* __restrict__ idx, const float* __restrict__ sc,
                         float* __restrict__ xk, float* __restrict__ Ak,
                         float* __restrict__ Mko, float* __restrict__ posk,
                         float* __restrict__ ssig, int Kin, int Kkeep, int pdim) {
  int t = blockIdx.x * blockDim.x + threadIdx.x;
  if (t >= BB * Kkeep) return;
  const int b = t / Kkeep, i = t % Kkeep;
  const int ri = idx[b * Kkeep + i];
  const float s = sc[b * Kkeep + i];
#pragma unroll
  for (int o = 0; o < 32; ++o)
    xk[(long long)t * 32 + o] = x[((long long)b * Kin + ri) * 32 + o] * s;
  if (posk != nullptr) {
    for (int j = 0; j < pdim; ++j)
      posk[(long long)t * pdim + j] = pos[((long long)b * Kin + ri) * pdim + j];
  }
  if (Ak != nullptr) {
    for (int j = 0; j < Kkeep; ++j) {
      const int rj = idx[b * Kkeep + j];
      Ak [((long long)b * Kkeep + i) * Kkeep + j] = A [((long long)b * Kin + ri) * Kin + rj];
      Mko[((long long)b * Kkeep + i) * Kkeep + j] = Mk[((long long)b * Kin + ri) * Kin + rj];
    }
  }
  ssig[t] = 1.0f / (1.0f + __expf(-s));
}

// A1 = A + I ; M1 = (M | eye) as float
__global__ void k_augpre(const float* __restrict__ Ak, const float* __restrict__ Mko,
                         float* __restrict__ A1, float* __restrict__ M1, int Kk) {
  long long t = (long long)blockIdx.x * blockDim.x + threadIdx.x;
  if (t >= (long long)BB * Kk * Kk) return;
  const int ij = (int)(t % (Kk * Kk));
  const int i = ij / Kk, j = ij % Kk;
  A1[t] = Ak[t] + ((i == j) ? 1.0f : 0.0f);
  M1[t] = ((Mko[t] != 0.0f) || (i == j)) ? 1.0f : 0.0f;
}

// A2 diag := 0 ; M2 = (M2f > 0) & ~eye
__global__ void k_augpost(float* __restrict__ A2, const float* __restrict__ M2f,
                          float* __restrict__ M2, int Kk) {
  long long t = (long long)blockIdx.x * blockDim.x + threadIdx.x;
  if (t >= (long long)BB * Kk * Kk) return;
  const int ij = (int)(t % (Kk * Kk));
  const int i = ij / Kk, j = ij % Kk;
  if (i == j) A2[t] = 0.0f;
  M2[t] = ((M2f[t] > 0.0f) && (i != j)) ? 1.0f : 0.0f;
}

// readout: out[b, off+o] = max_i xk[b,i,o] ; out[b, off+32+o] = mean_i
__global__ void k_readout(const float* __restrict__ xk, float* __restrict__ outb,
                          int Kkeep, int ldo, int off) {
  int t = blockIdx.x * blockDim.x + threadIdx.x;
  if (t >= BB * 32) return;
  const int b = t / 32, o = t % 32;
  float mx = -1e30f, sm = 0.0f;
  for (int i = 0; i < Kkeep; ++i) {
    const float v = xk[((long long)b * Kkeep + i) * 32 + o];
    mx = fmaxf(mx, v);
    sm += v;
  }
  outb[b * ldo + off + o]      = mx;
  outb[b * ldo + off + 32 + o] = sm / (float)Kkeep;
}

__global__ void k_fc_relu(const float* __restrict__ X, const float* __restrict__ W,
                          const float* __restrict__ bvec, float* __restrict__ Y,
                          int Mrows, int Kin, int Nout, int do_relu) {
  int t = blockIdx.x * blockDim.x + threadIdx.x;
  if (t >= Mrows * Nout) return;
  const int m = t / Nout, o = t % Nout;
  float a = bvec[o];
  for (int k = 0; k < Kin; ++k) a += X[(long long)m * Kin + k] * W[(long long)k * Nout + o];
  Y[t] = do_relu ? fmaxf(a, 0.0f) : a;
}

__global__ void k_bn(float* __restrict__ X, const float* __restrict__ g,
                     const float* __restrict__ bt, int Mrows, int Nout) {
  int o = blockIdx.x * blockDim.x + threadIdx.x;
  if (o >= Nout) return;
  float mean = 0.0f;
  for (int m = 0; m < Mrows; ++m) mean += X[(long long)m * Nout + o];
  mean /= (float)Mrows;
  float var = 0.0f;
  for (int m = 0; m < Mrows; ++m) {
    const float d = X[(long long)m * Nout + o] - mean;
    var += d * d;
  }
  var /= (float)Mrows;
  const float inv = rsqrtf(var + 1e-5f);
  for (int m = 0; m < Mrows; ++m)
    X[(long long)m * Nout + o] = (X[(long long)m * Nout + o] - mean) * inv * g[o] + bt[o];
}

__global__ void k_head(const float* __restrict__ X, const float* __restrict__ W,
                       const float* __restrict__ bvec, float* __restrict__ out) {
  int b = blockIdx.x * blockDim.x + threadIdx.x;
  if (b >= BB) return;
  float l0 = bvec[0], l1 = bvec[1];
  for (int k = 0; k < D3; ++k) {
    const float x = X[(long long)b * D3 + k];
    l0 += x * W[k * 2];
    l1 += x * W[k * 2 + 1];
  }
  const float mx = fmaxf(l0, l1);
  const float lse = mx + logf(__expf(l0 - mx) + __expf(l1 - mx));
  out[b * 2]     = l0 - lse;
  out[b * 2 + 1] = l1 - lse;
}

__global__ void k_copy(const float* __restrict__ src, float* __restrict__ dst, int n) {
  int t = blockIdx.x * blockDim.x + threadIdx.x;
  if (t < n) dst[t] = src[t];
}

// ---------------------------------------------------------------------------

extern "C" void kernel_launch(void* const* d_in, const int* in_sizes, int n_in,
                              void* d_out, int out_size, void* d_ws, size_t ws_size,
                              hipStream_t stream) {
  (void)in_sizes; (void)n_in; (void)out_size; (void)ws_size;

  const float* x_s     = (const float*)d_in[0];
  const int*   ei      = (const int*)  d_in[1];
  /* batch (d_in[2]) unused: graphs are uniform R-node blocks */
  const float* ea      = (const float*)d_in[3];
  const float* pos_s   = (const float*)d_in[4];
  const float* n1_w1   = (const float*)d_in[5];
  const float* n1_w2   = (const float*)d_in[6];
  const float* n1_b2   = (const float*)d_in[7];
  const float* conv1_b = (const float*)d_in[8];
  const float* pool1_w = (const float*)d_in[9];
  const float* n2_w1   = (const float*)d_in[10];
  const float* n2_w2   = (const float*)d_in[11];
  const float* n2_b2   = (const float*)d_in[12];
  const float* conv2_b = (const float*)d_in[13];
  const float* pool2_w = (const float*)d_in[14];
  const float* fc1_w   = (const float*)d_in[15];
  const float* fc1_b   = (const float*)d_in[16];
  const float* bn1_g   = (const float*)d_in[17];
  const float* bn1_b   = (const float*)d_in[18];
  const float* fc2_w   = (const float*)d_in[19];
  const float* fc2_b   = (const float*)d_in[20];
  const float* bn2_g   = (const float*)d_in[21];
  const float* bn2_b   = (const float*)d_in[22];
  const float* fc3_w   = (const float*)d_in[23];
  const float* fc3_b   = (const float*)d_in[24];

  float* out = (float*)d_out;
  // output: logp[128] | pool1_w[32] | pool2_w[32] | sig(s1)[6400] | sig(s2)[3200]
  float* out_logp = out;
  float* out_p1w  = out + 128;
  float* out_p2w  = out + 160;
  float* out_s1   = out + 192;
  float* out_s2   = out + 192 + BB * KP1;

  size_t off = 0;
  auto alloc = [&](size_t nbytes) -> void* {
    void* p = (char*)d_ws + off;
    off += ((nbytes + 255) / 256) * 256;
    return p;
  };
  auto allocf = [&](size_t n) -> float* { return (float*)alloc(n * 4); };
  auto allocs = [&](size_t n) -> unsigned short* { return (unsigned short*)alloc(n * 2); };
  auto cdiv = [](int a, int b) { return (a + b - 1) / b; };

  // f32 buffers
  float* adjA  = allocf((size_t)NNODES * RR);
  float* adjM  = allocf((size_t)NNODES * RR);     // contiguous with adjA
  float* W2r1  = allocf((size_t)INDIM * NCOLS);
  float* T1    = allocf((size_t)NNODES * NCOLS);
  float* wpos1 = allocf((size_t)NNODES * KCOMM);
  float* h1    = allocf((size_t)NNODES * D1);
  float* cv1   = allocf((size_t)NNODES * D1);
  float* sc1a  = allocf((size_t)NNODES);
  int*   idx1  = (int*)alloc((size_t)BB * KP1 * 4);
  float* sck1  = allocf((size_t)BB * KP1);
  float* xk1   = allocf((size_t)BB * KP1 * D1);
  float* Ak1   = allocf((size_t)BB * KP1 * KP1);
  float* Mk1   = allocf((size_t)BB * KP1 * KP1);
  float* posk  = allocf((size_t)BB * KP1 * RR);
  float* A1g   = allocf((size_t)BB * KP1 * KP1);
  float* M1g   = allocf((size_t)BB * KP1 * KP1);
  float* A2    = allocf((size_t)BB * KP1 * KP1);
  float* M2f   = allocf((size_t)BB * KP1 * KP1);
  float* M2m   = allocf((size_t)BB * KP1 * KP1);
  float* W2r2  = allocf((size_t)D1 * NCOLS);
  float* T2    = allocf((size_t)BB * KP1 * NCOLS);
  float* wpos2 = allocf((size_t)BB * KP1 * KCOMM);
  float* h2    = allocf((size_t)BB * KP1 * D2);
  float* cv2   = allocf((size_t)BB * KP1 * D2);
  float* sc2a  = allocf((size_t)BB * KP1);
  int*   idx2  = (int*)alloc((size_t)BB * KP2 * 4);
  float* sck2  = allocf((size_t)BB * KP2);
  float* xk2   = allocf((size_t)BB * KP2 * D2);
  float* hbuf  = allocf((size_t)BB * 128);
  float* f1o   = allocf((size_t)BB * D2);
  float* f2o   = allocf((size_t)BB * D3);
  // packed bf16 operand buffers
  unsigned short* xs_pk   = allocs((size_t)NNODES * KPAD200);            // A: x_s
  unsigned short* W2r1t   = allocs((size_t)NCOLS * KPAD200);             // Bt: W2r1
  unsigned short* al1_pk  = allocs((size_t)BB * MPAD200 * KPAD200);      // A: alpha1
  unsigned short* h1t     = allocs((size_t)BB * D1 * KPAD200);           // Bt: h1
  unsigned short* A1gA    = allocs((size_t)BB * NPAD100 * KPAD100);      // A: A1g
  unsigned short* A1gB    = allocs((size_t)BB * NPAD100 * KPAD100);      // Bt: A1g
  unsigned short* M1gA    = allocs((size_t)BB * NPAD100 * KPAD100);
  unsigned short* M1gB    = allocs((size_t)BB * NPAD100 * KPAD100);
  unsigned short* xk1_pk  = allocs((size_t)BB * KP1 * D1);               // A: xk1 (K=32)
  unsigned short* W2r2t   = allocs((size_t)NCOLS * D1);                  // Bt: W2r2 (K=32)
  unsigned short* al2_pk  = allocs((size_t)BB * NPAD100 * KPAD100);      // A: alpha2
  unsigned short* h2t     = allocs((size_t)BB * D2 * KPAD100);           // Bt: h2

  // ---- dense adjacency (zeroed every call: deterministic) ----
  {
    const long long nz = 2LL * NNODES * RR;
    k_zero<<<(int)((nz + 255) / 256), 256, 0, stream>>>(adjA, nz);
    k_build_adj<<<cdiv(NNODES, 256), 256, 0, stream>>>(ei, ea, adjA, adjM);
  }

  // ---- conv1: weight-gen GEMM T1 = x_s @ W2r1 ----
  k_w2r<<<cdiv(INDIM * NCOLS, 256), 256, 0, stream>>>(n1_w2, n1_b2, W2r1, INDIM);
  k_packA<<<cdiv(NNODES * KPAD200, 256), 256, 0, stream>>>(
      x_s, xs_pk, NNODES, INDIM, INDIM, NNODES, KPAD200, 0, 0);
  k_packBT<<<cdiv(NCOLS * KPAD200, 256), 256, 0, stream>>>(
      W2r1, W2r1t, INDIM, NCOLS, NCOLS, NCOLS, KPAD200, 0, 0);
  k_posw<<<cdiv(NNODES, 256), 256, 0, stream>>>(pos_s, n1_w1, wpos1, NNODES, RR);
  wmma_gemm_pk<<<dim3(NCOLS / 16, NNODES / 16, 1), 32, 0, stream>>>(
      xs_pk, W2r1t, T1, nullptr, NNODES, NCOLS, KPAD200, NCOLS, 0, 0, 0);
  k_combine<<<cdiv(NNODES * D1, 256), 256, 0, stream>>>(T1, wpos1, h1, NNODES);

  // ---- conv1: aggregate cv1 = softmax(Aw) @ h1 + bias ----
  k_softmax_pk<<<cdiv(BB * MPAD200, 256), 256, 0, stream>>>(
      adjA, adjM, al1_pk, RR, MPAD200, KPAD200);
  k_packBT<<<cdiv(D1 * KPAD200, 256), 256, 0, stream>>>(
      h1, h1t, RR, D1, D1, D1, KPAD200, (long long)RR * D1, (long long)D1 * KPAD200);
  // note: packBT batched over graphs
  {
    dim3 g(cdiv(D1 * KPAD200, 256), 1, BB);
    k_packBT<<<g, 256, 0, stream>>>(h1, h1t, RR, D1, D1, D1, KPAD200,
                                    (long long)RR * D1, (long long)D1 * KPAD200);
  }
  wmma_gemm_pk<<<dim3(D1 / 16, MPAD200 / 16, BB), 32, 0, stream>>>(
      al1_pk, h1t, cv1, conv1_b, RR, D1, KPAD200, D1,
      (long long)MPAD200 * KPAD200, (long long)D1 * KPAD200, (long long)RR * D1);

  // ---- pool1 + readout1 ----
  k_score<<<cdiv(NNODES, 256), 256, 0, stream>>>(cv1, pool1_w, sc1a, NNODES);
  k_topk<<<BB, 256, 0, stream>>>(sc1a, idx1, sck1, RR, KP1);
  k_gather<<<cdiv(BB * KP1, 256), 256, 0, stream>>>(
      cv1, adjA, adjM, pos_s, idx1, sck1, xk1, Ak1, Mk1, posk, out_s1, RR, KP1, RR);
  k_readout<<<cdiv(BB * 32, 256), 256, 0, stream>>>(xk1, hbuf, KP1, 128, 0);

  // ---- graph augmentation: (A+I)@(A+I), structural mask product ----
  k_augpre<<<cdiv(BB * KP1 * KP1, 256), 256, 0, stream>>>(Ak1, Mk1, A1g, M1g, KP1);
  {
    dim3 gp(cdiv(NPAD100 * KPAD100, 256), 1, BB);
    const long long sF = (long long)KP1 * KP1, sP = (long long)NPAD100 * KPAD100;
    k_packA <<<gp, 256, 0, stream>>>(A1g, A1gA, KP1, KP1, KP1, NPAD100, KPAD100, sF, sP);
    k_packBT<<<gp, 256, 0, stream>>>(A1g, A1gB, KP1, KP1, KP1, NPAD100, KPAD100, sF, sP);
    k_packA <<<gp, 256, 0, stream>>>(M1g, M1gA, KP1, KP1, KP1, NPAD100, KPAD100, sF, sP);
    k_packBT<<<gp, 256, 0, stream>>>(M1g, M1gB, KP1, KP1, KP1, NPAD100, KPAD100, sF, sP);
    dim3 gg(NPAD100 / 16, NPAD100 / 16, BB);
    wmma_gemm_pk<<<gg, 32, 0, stream>>>(A1gA, A1gB, A2, nullptr, KP1, KP1, KPAD100, KP1,
                                        sP, sP, sF);
    wmma_gemm_pk<<<gg, 32, 0, stream>>>(M1gA, M1gB, M2f, nullptr, KP1, KP1, KPAD100, KP1,
                                        sP, sP, sF);
  }
  k_augpost<<<cdiv(BB * KP1 * KP1, 256), 256, 0, stream>>>(A2, M2f, M2m, KP1);

  // ---- conv2 ----
  k_w2r<<<cdiv(D1 * NCOLS, 256), 256, 0, stream>>>(n2_w2, n2_b2, W2r2, D1);
  k_packBT<<<cdiv(NCOLS * D1, 256), 256, 0, stream>>>(
      W2r2, W2r2t, D1, NCOLS, NCOLS, NCOLS, D1, 0, 0);
  k_packA<<<cdiv(BB * KP1 * D1, 256), 256, 0, stream>>>(
      xk1, xk1_pk, BB * KP1, D1, D1, BB * KP1, D1, 0, 0);
  k_posw<<<cdiv(BB * KP1, 256), 256, 0, stream>>>(posk, n2_w1, wpos2, BB * KP1, RR);
  wmma_gemm_pk<<<dim3(NCOLS / 16, (BB * KP1) / 16, 1), 32, 0, stream>>>(
      xk1_pk, W2r2t, T2, nullptr, BB * KP1, NCOLS, D1, NCOLS, 0, 0, 0);
  k_combine<<<cdiv(BB * KP1 * D2, 256), 256, 0, stream>>>(T2, wpos2, h2, BB * KP1);

  k_softmax_pk<<<cdiv(BB * NPAD100, 256), 256, 0, stream>>>(
      A2, M2m, al2_pk, KP1, NPAD100, KPAD100);
  {
    dim3 g(cdiv(D2 * KPAD100, 256), 1, BB);
    k_packBT<<<g, 256, 0, stream>>>(h2, h2t, KP1, D2, D2, D2, KPAD100,
                                    (long long)KP1 * D2, (long long)D2 * KPAD100);
  }
  wmma_gemm_pk<<<dim3(D2 / 16, NPAD100 / 16, BB), 32, 0, stream>>>(
      al2_pk, h2t, cv2, conv2_b, KP1, D2, KPAD100, D2,
      (long long)NPAD100 * KPAD100, (long long)D2 * KPAD100, (long long)KP1 * D2);

  // ---- pool2 + readout2 ----
  k_score<<<cdiv(BB * KP1, 256), 256, 0, stream>>>(cv2, pool2_w, sc2a, BB * KP1);
  k_topk<<<BB, 256, 0, stream>>>(sc2a, idx2, sck2, KP1, KP2);
  k_gather<<<cdiv(BB * KP2, 256), 256, 0, stream>>>(
      cv2, nullptr, nullptr, nullptr, idx2, sck2, xk2, nullptr, nullptr, nullptr,
      out_s2, KP1, KP2, 0);
  k_readout<<<cdiv(BB * 32, 256), 256, 0, stream>>>(xk2, hbuf, KP2, 128, 64);

  // ---- MLP head ----
  k_fc_relu<<<cdiv(BB * D2, 256), 256, 0, stream>>>(hbuf, fc1_w, fc1_b, f1o, BB, 128, D2, 1);
  k_bn<<<cdiv(D2, 64), 64, 0, stream>>>(f1o, bn1_g, bn1_b, BB, D2);
  k_fc_relu<<<cdiv(BB * D3, 256), 256, 0, stream>>>(f1o, fc2_w, fc2_b, f2o, BB, D2, D3, 1);
  k_bn<<<cdiv(D3, 64), 64, 0, stream>>>(f2o, bn2_g, bn2_b, BB, D3);
  k_head<<<cdiv(BB, 64), 64, 0, stream>>>(f2o, fc3_w, fc3_b, out_logp);

  // ---- auxiliary outputs ----
  k_copy<<<1, 32, 0, stream>>>(pool1_w, out_p1w, 32);
  k_copy<<<1, 32, 0, stream>>>(pool2_w, out_p2w, 32);
}
